// SymbolicMultiHeadAttention_22342419874058
// MI455X (gfx1250) — matrix operand order repaired and checked by reading, not compiled, not run on hardware
//
#include <hip/hip_runtime.h>
#include <stdint.h>

#define B_   8
#define T_   512
#define V_   2048
#define H_   4
#define NE_  768
#define VH_  8192   // V*H   (cooc_flat row length)
#define C3_  3072   // H*NE  (Wtok row length)
#define K2_  2048   // T*H   (GEMM-2 K)
#define LDT  34     // padded LDS stride (bf16 elems); 17 banks, coprime with 64

typedef __attribute__((ext_vector_type(16))) __bf16        v16bf;
typedef __attribute__((ext_vector_type(8)))  float         v8f;
typedef __attribute__((ext_vector_type(8)))  unsigned int  v8u;

union FragB { v8u u; v16bf b; };

__device__ __forceinline__ unsigned short f2bf(float f) {
  unsigned u = __builtin_bit_cast(unsigned, f);
  u += 0x7FFFu + ((u >> 16) & 1u);         // round-to-nearest-even
  return (unsigned short)(u >> 16);
}
__device__ __forceinline__ unsigned pack2(float lo, float hi) {
  return (unsigned)f2bf(lo) | ((unsigned)f2bf(hi) << 16);
}

// --- CDNA5 async global->LDS copy (ASYNCcnt path) --------------------------
__device__ __forceinline__ void async_copy_b128(unsigned ldsOff, const void* g) {
  // LDS address comes from the VDST operand;  16 bytes per lane, completes asynchronously
  asm volatile("global_load_async_to_lds_b128 %0, %1, off"
               :: "v"(ldsOff), "v"(g) : "memory");
}
__device__ __forceinline__ void wait_async0() {
#if __has_builtin(__builtin_amdgcn_s_wait_asynccnt)
  __builtin_amdgcn_s_wait_asynccnt(0);
#else
  asm volatile("s_wait_asynccnt 0" ::: "memory");
#endif
}

// ---------------------------------------------------------------------------
// Kernel 1:  Wtok[v, h*NE+n] = sum_d bf16(cooc[v,d]) * bf16(proj_w[n, h*VH+d])
// M=2048, N=3072, K=8192.  128x128 block tile, 8 waves (4x2), K-step 32,
// double-buffered LDS with register staging + prefetch 2 steps ahead.
// ---------------------------------------------------------------------------
__global__ __launch_bounds__(256) void k_gemm_wtok(
    const float* __restrict__ cooc,
    const float* __restrict__ projw,
    unsigned short* __restrict__ wtok)
{
  __shared__ __align__(16) unsigned short As[2][128 * LDT];
  __shared__ __align__(16) unsigned short Bs[2][128 * LDT];

  const int tid  = threadIdx.x;
  const int m0   = blockIdx.y * 128;
  const int c0   = blockIdx.x * 128;
  const int wave = tid >> 5, lane = tid & 31;
  const int wrow = wave & 3, wcol = wave >> 2;   // 4 waves M, 2 waves N
  const int m16  = lane & 15, half = lane >> 4;

  // cooperative loader coordinates: 2 threads per row/col, 16 K-elems each
  const int lr = tid >> 1;
  const int ks = (tid & 1) * 16;

  // B column mapping (column c -> proj_w row n, h block)
  const int cg = c0 + lr;
  const float* gaBase = cooc + (long long)(m0 + lr) * VH_ + ks;
  const float* gbBase = projw + (long long)(cg % NE_) * (VH_ * H_)
                              + (long long)(cg / NE_) * VH_ + ks;

  float4 sa[4], sb[4];
  auto stage_load = [&](int k0) {
    const float4* ga = (const float4*)(gaBase + k0);
    const float4* gb = (const float4*)(gbBase + k0);
#pragma unroll
    for (int q = 0; q < 4; q++) { sa[q] = ga[q]; sb[q] = gb[q]; }
  };
  auto stage_store = [&](int buf) {
    unsigned* da = (unsigned*)&As[buf][lr * LDT + ks];
    unsigned* db = (unsigned*)&Bs[buf][lr * LDT + ks];
#pragma unroll
    for (int q = 0; q < 4; q++) {
      da[2*q]   = pack2(sa[q].x, sa[q].y);
      da[2*q+1] = pack2(sa[q].z, sa[q].w);
      db[2*q]   = pack2(sb[q].x, sb[q].y);
      db[2*q+1] = pack2(sb[q].z, sb[q].w);
    }
  };

  v8f acc[2][4] = {};

  stage_load(0);
  stage_store(0);
  __syncthreads();

  int cur = 0;
  for (int k0 = 0; k0 < VH_; k0 += 32) {
    const bool more = (k0 + 32) < VH_;
    if (more) {
      stage_load(k0 + 32);
      if ((k0 + 64) < VH_) {                       // warm L2/L0 two steps ahead
        __builtin_prefetch(gaBase + k0 + 64, 0, 3);
        __builtin_prefetch(gbBase + k0 + 64, 0, 3);
      }
    }

    FragB afrag[2], bfrag[4];
#pragma unroll
    for (int i = 0; i < 2; i++) {
      const unsigned* src = (const unsigned*)&As[cur][(wrow*32 + i*16 + m16) * LDT];
#pragma unroll
      for (int r = 0; r < 8; r++)                  // 16-bit A 16x32 VGPR layout
        afrag[i].u[r] = src[((r < 4) ? 0 : 8) + half*4 + (r & 3)];
    }
#pragma unroll
    for (int j = 0; j < 4; j++) {
      const unsigned* src = (const unsigned*)&Bs[cur][(wcol*64 + j*16 + m16) * LDT];
#pragma unroll
      for (int r = 0; r < 8; r++)                  // 16-bit B 32x16 VGPR layout
        bfrag[j].u[r] = src[((r < 4) ? 0 : 8) + half*4 + (r & 3)];
    }
#pragma unroll
    for (int i = 0; i < 2; i++)
#pragma unroll
      for (int j = 0; j < 4; j++)
        acc[i][j] = __builtin_amdgcn_wmma_f32_16x16x32_bf16(
            false, afrag[i].b, false, bfrag[j].b, (short)0, acc[i][j], false, false);

    if (more) { stage_store(cur ^ 1); cur ^= 1; }
    __syncthreads();
  }

#pragma unroll
  for (int i = 0; i < 2; i++)
#pragma unroll
    for (int j = 0; j < 4; j++) {
      const int n = c0 + wcol*64 + j*16 + m16;
#pragma unroll
      for (int g = 0; g < 8; g++) {
        const int m = m0 + wrow*32 + i*16 + g + half*8;   // C layout: M = g + 8*half
        wtok[(long long)m * C3_ + n] = f2bf(acc[i][j][g]);
      }
    }
}

// ---------------------------------------------------------------------------
// Kernel 2: att[b,i,j,h] = softmax_j(cooc[tok_i,tok_j,h]) + (i==j), bf16.
// One block per (b,i); 256 threads cover j in {tid, tid+256}; H=4 via float4.
// ---------------------------------------------------------------------------
__global__ __launch_bounds__(256) void k_softmax(
    const int* __restrict__ x,
    const float* __restrict__ cooc,
    unsigned short* __restrict__ att)
{
  __shared__ int    tok_s[T_];
  __shared__ float4 red[256];

  const int tid = threadIdx.x;
  const int b   = blockIdx.x >> 9;
  const int i   = blockIdx.x & (T_ - 1);

  for (int j = tid; j < T_; j += 256) tok_s[j] = x[b * T_ + j];
  __syncthreads();

  const float4* row = (const float4*)cooc + (long long)tok_s[i] * (VH_ / H_);
  const int j0 = tid, j1 = tid + 256;
  const float4 w0 = row[tok_s[j0]];
  const float4 w1 = row[tok_s[j1]];

  float4 r;
  r.x = fmaxf(w0.x, w1.x); r.y = fmaxf(w0.y, w1.y);
  r.z = fmaxf(w0.z, w1.z); r.w = fmaxf(w0.w, w1.w);
  red[tid] = r; __syncthreads();
  for (int s = 128; s; s >>= 1) {
    if (tid < s) {
      float4 a = red[tid], c = red[tid + s];
      a.x = fmaxf(a.x, c.x); a.y = fmaxf(a.y, c.y);
      a.z = fmaxf(a.z, c.z); a.w = fmaxf(a.w, c.w);
      red[tid] = a;
    }
    __syncthreads();
  }
  const float4 mx = red[0];
  __syncthreads();

  float4 e0, e1;
  e0.x = __expf(w0.x - mx.x); e0.y = __expf(w0.y - mx.y);
  e0.z = __expf(w0.z - mx.z); e0.w = __expf(w0.w - mx.w);
  e1.x = __expf(w1.x - mx.x); e1.y = __expf(w1.y - mx.y);
  e1.z = __expf(w1.z - mx.z); e1.w = __expf(w1.w - mx.w);

  r.x = e0.x + e1.x; r.y = e0.y + e1.y; r.z = e0.z + e1.z; r.w = e0.w + e1.w;
  red[tid] = r; __syncthreads();
  for (int s = 128; s; s >>= 1) {
    if (tid < s) {
      float4 a = red[tid], c = red[tid + s];
      a.x += c.x; a.y += c.y; a.z += c.z; a.w += c.w;
      red[tid] = a;
    }
    __syncthreads();
  }
  const float4 sm = red[0];
  float4 inv; inv.x = 1.f/sm.x; inv.y = 1.f/sm.y; inv.z = 1.f/sm.z; inv.w = 1.f/sm.w;

  float4 a0, a1;
  a0.x = e0.x*inv.x; a0.y = e0.y*inv.y; a0.z = e0.z*inv.z; a0.w = e0.w*inv.w;
  a1.x = e1.x*inv.x; a1.y = e1.y*inv.y; a1.z = e1.z*inv.z; a1.w = e1.w*inv.w;
  if (j0 == i) { a0.x += 1.f; a0.y += 1.f; a0.z += 1.f; a0.w += 1.f; }
  if (j1 == i) { a1.x += 1.f; a1.y += 1.f; a1.z += 1.f; a1.w += 1.f; }

  uint2* dst = (uint2*)att + (long long)(b * T_ + i) * T_;
  dst[j0] = make_uint2(pack2(a0.x, a0.y), pack2(a0.z, a0.w));
  dst[j1] = make_uint2(pack2(a1.x, a1.y), pack2(a1.z, a1.w));
}

// ---------------------------------------------------------------------------
// Kernel 3: per b:  Y(512x768) = Att(512x2048) @ P(2048x768),
// P[j*4+h, n] = Wtok[tok[b,j], h*NE + n] gathered during LDS load. + bias.
// A-tile via async global->LDS (ASYNCcnt), B-tile staged gather/transpose,
// double-buffered LDS, prefetch, non-temporal output stores.
// ---------------------------------------------------------------------------
__global__ __launch_bounds__(256) void k_gemm_out(
    const int* __restrict__ x,
    const unsigned short* __restrict__ att,
    const unsigned short* __restrict__ wtok,
    const float* __restrict__ projb,
    float* __restrict__ out)
{
  __shared__ __align__(16) unsigned short As[2][128 * LDT];
  __shared__ __align__(16) unsigned short Bs[2][128 * LDT];
  __shared__ int tok_s[T_];

  const int tid = threadIdx.x;
  const int b   = blockIdx.z;
  const int m0  = blockIdx.y * 128;
  const int c0  = blockIdx.x * 128;
  const int wave = tid >> 5, lane = tid & 31;
  const int wrow = wave & 3, wcol = wave >> 2;
  const int m16  = lane & 15, half = lane >> 4;

  for (int j = tid; j < T_; j += 256) tok_s[j] = x[b * T_ + j];
  __syncthreads();

  const unsigned* attb = (const unsigned*)(att + (long long)b * T_ * K2_);

  const int lr = tid >> 1;          // A loader: row, 2 threads/row
  const int ks = (tid & 1) * 16;
  const int kl = tid & 31;          // B loader: K lane
  const int cs = (tid >> 5) * 16;   // B loader: 16-col group

  // A tile: pure bf16 copy -> async global->LDS, 2x b128 per thread
  auto a_async = [&](int k0, int buf) {
    const void* g = attb + (((m0 + lr) * K2_ + k0 + ks) >> 1);
    const unsigned ldsOff = (unsigned)(size_t)&As[buf][lr * LDT + ks];
    async_copy_b128(ldsOff,      g);
    async_copy_b128(ldsOff + 16, (const char*)g + 16);
  };

  unsigned sb[8];
  auto b_stage_load = [&](int k0) {
    const int k = k0 + kl;
    const unsigned* gb =
        (const unsigned*)(wtok + (long long)tok_s[k >> 2] * C3_ + (k & 3) * NE_ + (c0 + cs));
#pragma unroll
    for (int q = 0; q < 8; q++) sb[q] = gb[q];
  };
  auto b_stage_store = [&](int buf) {
#pragma unroll
    for (int q = 0; q < 16; q++)
      Bs[buf][(cs + q) * LDT + kl] =
          (unsigned short)((q & 1) ? (sb[q >> 1] >> 16) : (sb[q >> 1] & 0xFFFFu));
  };

  v8f acc[2][4] = {};

  a_async(0, 0);
  b_stage_load(0);
  b_stage_store(0);
  wait_async0();
  __syncthreads();

  int cur = 0;
  for (int k0 = 0; k0 < K2_; k0 += 32) {
    const bool more = (k0 + 32) < K2_;
    if (more) {
      a_async(k0 + 32, cur ^ 1);      // async engine fills next buffer during WMMAs
      b_stage_load(k0 + 32);
      if ((k0 + 64) < K2_) {
        const int kp = k0 + 64 + kl;
        __builtin_prefetch(
            wtok + (long long)tok_s[kp >> 2] * C3_ + (kp & 3) * NE_ + (c0 + cs), 0, 3);
      }
    }

    FragB afrag[2], bfrag[4];
#pragma unroll
    for (int i = 0; i < 2; i++) {
      const unsigned* src = (const unsigned*)&As[cur][(wrow*32 + i*16 + m16) * LDT];
#pragma unroll
      for (int r = 0; r < 8; r++)
        afrag[i].u[r] = src[((r < 4) ? 0 : 8) + half*4 + (r & 3)];
    }
#pragma unroll
    for (int j = 0; j < 4; j++) {
      const unsigned* src = (const unsigned*)&Bs[cur][(wcol*64 + j*16 + m16) * LDT];
#pragma unroll
      for (int r = 0; r < 8; r++)
        bfrag[j].u[r] = src[((r < 4) ? 0 : 8) + half*4 + (r & 3)];
    }
#pragma unroll
    for (int i = 0; i < 2; i++)
#pragma unroll
      for (int j = 0; j < 4; j++)
        acc[i][j] = __builtin_amdgcn_wmma_f32_16x16x32_bf16(
            false, afrag[i].b, false, bfrag[j].b, (short)0, acc[i][j], false, false);

    if (more) { b_stage_store(cur ^ 1); cur ^= 1; }
    wait_async0();                    // async A-tile for next step must be in LDS
    __syncthreads();
  }

#pragma unroll
  for (int j = 0; j < 4; j++) {
    const int n = c0 + wcol*64 + j*16 + m16;
    const float bias = projb[n];
#pragma unroll
    for (int i = 0; i < 2; i++)
#pragma unroll
      for (int g = 0; g < 8; g++) {
        const int m = m0 + wrow*32 + i*16 + g + half*8;
        __builtin_nontemporal_store(acc[i][j][g] + bias,
                                    out + (long long)(b * T_ + m) * NE_ + n);
      }
  }
}

// ---------------------------------------------------------------------------
extern "C" void kernel_launch(void* const* d_in, const int* in_sizes, int n_in,
                              void* d_out, int out_size, void* d_ws, size_t ws_size,
                              hipStream_t stream) {
  const int*   x     = (const int*)d_in[0];
  const float* cooc  = (const float*)d_in[1];
  const float* projw = (const float*)d_in[2];
  const float* projb = (const float*)d_in[3];
  float*       out   = (float*)d_out;

  // workspace: att bf16 (16.8 MB) + Wtok bf16 (12.6 MB)
  unsigned short* att  = (unsigned short*)d_ws;
  unsigned short* wtok = att + (size_t)B_ * T_ * T_ * H_;

  k_gemm_wtok<<<dim3(C3_ / 128, V_ / 128), 256, 0, stream>>>(cooc, projw, wtok);
  k_softmax  <<<dim3(B_ * T_),             256, 0, stream>>>(x, cooc, att);
  k_gemm_out <<<dim3(NE_ / 128, T_ / 128, B_), 256, 0, stream>>>(x, att, wtok, projb, out);

  (void)in_sizes; (void)n_in; (void)out_size; (void)ws_size;
}
